// MultiHeadAttention_83614423318615
// MI455X (gfx1250) — matrix operand order repaired and checked by reading, not compiled, not run on hardware
//
#include <hip/hip_runtime.h>

// Problem constants (match reference)
#define B_   2
#define Tq   2048
#define Cc   2048
#define NH   16
#define NKV  4
#define HDm  128

typedef __attribute__((ext_vector_type(16))) __bf16 v16bf;
typedef __attribute__((ext_vector_type(8)))  __bf16 v8bf;
typedef __attribute__((ext_vector_type(4)))  __bf16 v4bf;
typedef __attribute__((ext_vector_type(8)))  float  v8f;
typedef __attribute__((ext_vector_type(4)))  unsigned int u32x4;
typedef __attribute__((ext_vector_type(8)))  int          i32x8;
typedef __attribute__((ext_vector_type(4)))  int          i32x4;

#if defined(__has_builtin)
#  if __has_builtin(__builtin_amdgcn_tensor_load_to_lds) && __has_builtin(__builtin_amdgcn_s_wait_tensorcnt)
#    define HAVE_TDM 1
#  endif
#endif
#ifndef HAVE_TDM
#  define HAVE_TDM 0
#endif

__device__ __forceinline__ v16bf mk16(v8bf a, v8bf b) {
  return __builtin_shufflevector(a, b, 0,1,2,3,4,5,6,7,8,9,10,11,12,13,14,15);
}

#if HAVE_TDM
// ---------------------------------------------------------------------------
// TDM: DMA a 2D bf16 tile (tile_h rows x tile_w elems, row stride = stride
// elems) from global to LDS (rows packed contiguously). D# per ISA ch.8.
// ---------------------------------------------------------------------------
__device__ __forceinline__ void tdm_load_tile(unsigned lds_off, const void* gaddr,
                                              int tile_w, int tile_h, int stride) {
  unsigned long long ga = (unsigned long long)(uintptr_t)gaddr;
  u32x4 g0;
  g0[0] = 1u;                                           // count=1, user descriptor
  g0[1] = lds_off;                                      // lds_addr (bytes)
  g0[2] = (unsigned)(ga & 0xffffffffu);                 // global_addr[31:0]
  g0[3] = (unsigned)((ga >> 32) & 0x01ffffffu)          // global_addr[56:32]
        | (2u << 30);                                   // type = 2 ("image")
  i32x8 g1;
  g1[0] = (1 << 16);                                    // data_size=1 -> 2 bytes
  g1[1] = (tile_w & 0xffff) << 16;                      // tensor_dim0[15:0]   (=tile_w, exact fit)
  g1[2] = ((tile_w >> 16) & 0xffff)                     // tensor_dim0[31:16]
        | ((tile_h & 0xffff) << 16);                    // tensor_dim1[15:0]
  g1[3] = ((tile_h >> 16) & 0xffff)                     // tensor_dim1[31:16]
        | ((tile_w & 0xffff) << 16);                    // tile_dim0
  g1[4] = (tile_h & 0xffff);                            // tile_dim1 (tile_dim2=0)
  g1[5] = stride;                                       // tensor_dim0_stride[31:0]
  g1[6] = 0;                                            // stride hi / dim1_stride
  g1[7] = 0;
  i32x4 gz = {0, 0, 0, 0};
#if __clang_major__ >= 23
  i32x8 gz8 = {0, 0, 0, 0, 0, 0, 0, 0};
  __builtin_amdgcn_tensor_load_to_lds(g0, g1, gz, gz, gz8, 0);
#else
  __builtin_amdgcn_tensor_load_to_lds(g0, g1, gz, gz, 0);
#endif
}
#endif

// ---------------------------------------------------------------------------
// fp32 -> bf16 (vectorized x4)
// ---------------------------------------------------------------------------
__global__ __launch_bounds__(256) void cvt_f32_bf16(const float* __restrict__ in,
                                                    __bf16* __restrict__ out, int n4) {
  int i = blockIdx.x * 256 + threadIdx.x;
  if (i >= n4) return;
  float4 v = ((const float4*)in)[i];
  v4bf o;
  o[0] = (__bf16)v.x; o[1] = (__bf16)v.y; o[2] = (__bf16)v.z; o[3] = (__bf16)v.w;
  ((v4bf*)out)[i] = o;
}

// ---------------------------------------------------------------------------
// fp32 (Kd x Nd, row-major) -> bf16 transposed (Nd x Kd, K-contiguous)
// ---------------------------------------------------------------------------
__global__ __launch_bounds__(256) void cvtT_f32_bf16(const float* __restrict__ in,
                                                     __bf16* __restrict__ out,
                                                     int Kd, int Nd) {
  __shared__ float tile[32][33];
  int k0 = blockIdx.x * 32, n0 = blockIdx.y * 32;
  int tx = threadIdx.x & 31, ty = threadIdx.x >> 5;
  for (int r = ty; r < 32; r += 8)
    tile[r][tx] = in[(size_t)(k0 + r) * Nd + n0 + tx];
  __syncthreads();
  for (int r = ty; r < 32; r += 8)
    out[(size_t)(n0 + r) * Kd + k0 + tx] = (__bf16)tile[tx][r];
}

// ---------------------------------------------------------------------------
// bf16 WMMA GEMM: C(MxN) = A(MxK, row-major) * Bt(NxK, row-major)^T
// 256 thr = 8 waves; block tile 128x128; wave tile 32x64; K step 32.
// TDM path: wave 0 DMAs A/B tiles into double-buffered LDS (s_wait_tensorcnt),
// all waves read WMMA fragments from LDS.
// ---------------------------------------------------------------------------
template <bool OUT_BF16>
__global__ __launch_bounds__(256) void gemm_bf16_k(const __bf16* __restrict__ A,
                                                   const __bf16* __restrict__ Bt,
                                                   void* __restrict__ Cout,
                                                   int M, int N, int K) {
  const int lane = threadIdx.x & 31;
  const int wid  = threadIdx.x >> 5;
  const int wr = wid & 3, wc = wid >> 2;
  const int lm = lane & 15;
  const bool hi = (lane & 16) != 0;
  const int kbA = hi ? 8 : 0;    // A frag: K runs [kbA,kbA+8) and [kbA+16,kbA+24)
  const int kbB = hi ? 8 : 0;    // B frag: K runs [kbB,kbB+8) and [kbB+16,kbB+24)

  v8f z = {};
  v8f acc[2][4];
#pragma unroll
  for (int i = 0; i < 2; ++i)
#pragma unroll
    for (int j = 0; j < 4; ++j) acc[i][j] = z;

#if HAVE_TDM
  __shared__ __bf16 As[2][128][32];   // 8 KB x2
  __shared__ __bf16 Bs[2][128][32];   // 8 KB x2
  const __bf16* Ablk = A  + (size_t)blockIdx.x * 128 * K;
  const __bf16* Bblk = Bt + (size_t)blockIdx.y * 128 * K;

  if (wid == 0) {
    tdm_load_tile((unsigned)(uintptr_t)&As[0][0][0], Ablk, 32, 128, K);
    tdm_load_tile((unsigned)(uintptr_t)&Bs[0][0][0], Bblk, 32, 128, K);
  }
  int buf = 0;
  for (int k0 = 0; k0 < K; k0 += 32) {
    if (wid == 0) {
      if (k0 + 32 < K) {
        tdm_load_tile((unsigned)(uintptr_t)&As[buf ^ 1][0][0], Ablk + k0 + 32, 32, 128, K);
        tdm_load_tile((unsigned)(uintptr_t)&Bs[buf ^ 1][0][0], Bblk + k0 + 32, 32, 128, K);
        __builtin_amdgcn_s_wait_tensorcnt(2);   // current pair landed
      } else {
        __builtin_amdgcn_s_wait_tensorcnt(0);   // last pair landed
      }
    }
    __syncthreads();
    v16bf af[2], bf2[4];
#pragma unroll
    for (int i = 0; i < 2; ++i) {
      const __bf16* p = &As[buf][wr * 32 + i * 16 + lm][kbA];
      af[i] = mk16(*(const v8bf*)p, *(const v8bf*)(p + 16));
    }
#pragma unroll
    for (int j = 0; j < 4; ++j) {
      const __bf16* p = &Bs[buf][wc * 64 + j * 16 + lm][kbB];
      bf2[j] = mk16(*(const v8bf*)p, *(const v8bf*)(p + 16));
    }
#pragma unroll
    for (int i = 0; i < 2; ++i)
#pragma unroll
      for (int j = 0; j < 4; ++j)
        acc[i][j] = __builtin_amdgcn_wmma_f32_16x16x32_bf16(
            false, af[i], false, bf2[j], (short)0, acc[i][j], false, false);
    __syncthreads();
    buf ^= 1;
  }
#else
  // Fallback: fragments straight from global (b128 pairs).
  const int m0f = blockIdx.x * 128 + wr * 32;
  const int n0f = blockIdx.y * 128 + wc * 64;
  const __bf16* pa0 = A + (size_t)(m0f + lm) * K + kbA;
  const __bf16* pa1 = A + (size_t)(m0f + 16 + lm) * K + kbA;
  const __bf16* pb0 = Bt + (size_t)(n0f + lm) * K + kbB;
  for (int k0 = 0; k0 < K; k0 += 32) {
    v16bf af[2];
    af[0] = mk16(*(const v8bf*)(pa0 + k0), *(const v8bf*)(pa0 + k0 + 16));
    af[1] = mk16(*(const v8bf*)(pa1 + k0), *(const v8bf*)(pa1 + k0 + 16));
    v16bf bf2[4];
#pragma unroll
    for (int j = 0; j < 4; ++j) {
      const __bf16* q = pb0 + (size_t)j * 16 * K + k0;
      bf2[j] = mk16(*(const v8bf*)q, *(const v8bf*)(q + 16));
    }
#pragma unroll
    for (int i = 0; i < 2; ++i)
#pragma unroll
      for (int j = 0; j < 4; ++j)
        acc[i][j] = __builtin_amdgcn_wmma_f32_16x16x32_bf16(
            false, af[i], false, bf2[j], (short)0, acc[i][j], false, false);
  }
#endif

  const int m0 = blockIdx.x * 128 + wr * 32;
  const int n0 = blockIdx.y * 128 + wc * 64;
  const int mo = hi ? 8 : 0;
#pragma unroll
  for (int i = 0; i < 2; ++i)
#pragma unroll
    for (int j = 0; j < 4; ++j)
#pragma unroll
      for (int r = 0; r < 8; ++r) {
        size_t row = (size_t)(m0 + i * 16 + r + mo);
        size_t col = (size_t)(n0 + j * 16 + lm);
        if constexpr (OUT_BF16)
          ((__bf16*)Cout)[row * N + col] = (__bf16)acc[i][j][r];
        else
          ((float*)Cout)[row * N + col] = acc[i][j][r];
      }
}

// ---------------------------------------------------------------------------
// RoPE in place on bf16 (B, Tq, H, HDm); one thread per (b,t,h,pair)
// ---------------------------------------------------------------------------
__global__ __launch_bounds__(256) void rope_bf16(__bf16* __restrict__ q,
                                                 const float* __restrict__ cs,
                                                 const float* __restrict__ sn,
                                                 int H, int total) {
  int idx = blockIdx.x * 256 + threadIdx.x;
  if (idx >= total) return;
  const int half = HDm / 2;
  int p = idx % half; int r = idx / half;
  int h = r % H; r /= H;
  int t = r % Tq; int b = r / Tq;
  float c = cs[t * half + p], s = sn[t * half + p];
  size_t base = (((size_t)(b * Tq + t) * H + h) * HDm) + 2 * p;
  float e = (float)q[base], o = (float)q[base + 1];
  q[base]     = (__bf16)(e * c - o * s);
  q[base + 1] = (__bf16)(e * s + o * c);
}

// ---------------------------------------------------------------------------
// V (B, Tq, NKV, HDm) -> Vt (B, NKV, HDm, Tq)
// ---------------------------------------------------------------------------
__global__ __launch_bounds__(256) void transpose_v(const __bf16* __restrict__ v,
                                                   __bf16* __restrict__ vt) {
  __shared__ __bf16 tile[32][33];
  int t0 = blockIdx.x * 32, d0 = blockIdx.y * 32;
  int bk = blockIdx.z; int kv = bk % NKV; int b = bk / NKV;
  int tx = threadIdx.x & 31, ty = threadIdx.x >> 5;
  for (int r = ty; r < 32; r += 8)
    tile[r][tx] = v[((size_t)(b * Tq + t0 + r) * NKV + kv) * HDm + d0 + tx];
  __syncthreads();
  for (int r = ty; r < 32; r += 8)
    vt[(((size_t)(b * NKV + kv) * HDm) + d0 + r) * Tq + t0 + tx] = tile[tx][r];
}

// ---------------------------------------------------------------------------
// Flash attention (causal, GQA). Block = 128 thr = 4 waves; each wave owns 16
// query rows (one M-frag). Q A-frags live in registers; K/Vt B-frags loaded
// directly from global; P relayout (D->A) via per-wave LDS tile.
// grid = (Tq/64, B*NH)
// ---------------------------------------------------------------------------
__global__ __launch_bounds__(128) void attn_flash(const __bf16* __restrict__ Q,
                                                  const __bf16* __restrict__ Km,
                                                  const __bf16* __restrict__ Vt,
                                                  __bf16* __restrict__ O) {
  const int lane = threadIdx.x & 31;
  const int wid  = threadIdx.x >> 5;
  const int lm = lane & 15;
  const bool hi = (lane & 16) != 0;
  const int kbA = hi ? 8 : 0;
  const int kbB = hi ? 8 : 0;
  const int mo  = hi ? 8 : 0;

  const int bh = blockIdx.y;
  const int b  = bh / NH;
  const int h  = bh % NH;
  const int kv = h / (NH / NKV);
  const int q0 = blockIdx.x * 64 + wid * 16;

  __shared__ __bf16 pl[4][16][32];   // per-wave P tile (16 rows x 32 keys)

  // Q A-fragments for the full HD=128 (4 K-steps), kept in registers
  v16bf aq[4];
  {
    const __bf16* qb = Q + ((size_t)(b * Tq + q0 + lm) * NH + h) * HDm;
#pragma unroll
    for (int ks = 0; ks < 4; ++ks) {
      const __bf16* p = qb + ks * 32 + kbA;
      aq[ks] = mk16(*(const v8bf*)p, *(const v8bf*)(p + 16));
    }
  }

  v8f z = {};
  v8f o[8];
  float m[8], l[8];
#pragma unroll
  for (int f = 0; f < 8; ++f) o[f] = z;
#pragma unroll
  for (int r = 0; r < 8; ++r) { m[r] = -1e30f; l[r] = 0.0f; }

  const float scale = 0.08838834764831845f;   // 1/sqrt(128)

  const int kend = q0 + 16;                   // causal: keys < q0+16
  for (int kb = 0; kb < kend; kb += 32) {
    // ---- S = Q K^T  (16 x 32 over this key block) ----
    v8f s[2]; s[0] = z; s[1] = z;
#pragma unroll
    for (int ks = 0; ks < 4; ++ks) {
#pragma unroll
      for (int j = 0; j < 2; ++j) {
        const __bf16* p = Km + ((size_t)(b * Tq + kb + j * 16 + lm) * NKV + kv) * HDm
                        + ks * 32 + kbB;
        v16bf bk = mk16(*(const v8bf*)p, *(const v8bf*)(p + 16));
        s[j] = __builtin_amdgcn_wmma_f32_16x16x32_bf16(
            false, aq[ks], false, bk, (short)0, s[j], false, false);
      }
    }
    __builtin_prefetch(Km + ((size_t)(b * Tq + kb + 32 + lm) * NKV + kv) * HDm, 0, 0);
    // ---- online softmax (row = r+mo per VGPR, cols across 16 lanes) ----
#pragma unroll
    for (int r = 0; r < 8; ++r) {
      const int qi = q0 + r + mo;
      const int k0j = kb + lm, k1j = kb + 16 + lm;
      float v0 = (k0j <= qi) ? s[0][r] * scale : -1e30f;
      float v1 = (k1j <= qi) ? s[1][r] * scale : -1e30f;
      float rm = fmaxf(v0, v1);
#pragma unroll
      for (int off = 1; off < 16; off <<= 1)
        rm = fmaxf(rm, __shfl_xor(rm, off, 16));
      float mn = fmaxf(m[r], rm);
      float p0 = __expf(v0 - mn);
      float p1 = __expf(v1 - mn);
      float rs = p0 + p1;
#pragma unroll
      for (int off = 1; off < 16; off <<= 1)
        rs += __shfl_xor(rs, off, 16);
      float alpha = __expf(m[r] - mn);
      l[r] = l[r] * alpha + rs;
      m[r] = mn;
#pragma unroll
      for (int f = 0; f < 8; ++f) o[f][r] *= alpha;
      pl[wid][r + mo][lm]      = (__bf16)p0;
      pl[wid][r + mo][16 + lm] = (__bf16)p1;
    }
    // ---- reload P as A-fragment (D-layout -> A-layout via LDS) ----
    const __bf16* pp = &pl[wid][lm][kbA];
    v16bf ap = mk16(*(const v8bf*)pp, *(const v8bf*)(pp + 16));
    // ---- O += P V ----
#pragma unroll
    for (int f = 0; f < 8; ++f) {
      const __bf16* pv = Vt + (((size_t)(b * NKV + kv) * HDm) + f * 16 + lm) * Tq
                       + kb + kbB;
      v16bf bv = mk16(*(const v8bf*)pv, *(const v8bf*)(pv + 16));
      o[f] = __builtin_amdgcn_wmma_f32_16x16x32_bf16(
          false, ap, false, bv, (short)0, o[f], false, false);
    }
  }

  // ---- normalize and store ----
  float inv[8];
#pragma unroll
  for (int r = 0; r < 8; ++r) inv[r] = 1.0f / l[r];
#pragma unroll
  for (int f = 0; f < 8; ++f)
#pragma unroll
    for (int r = 0; r < 8; ++r) {
      int t = q0 + r + mo;
      O[((size_t)(b * Tq + t) * NH + h) * HDm + f * 16 + lm] =
          (__bf16)(o[f][r] * inv[r]);
    }
}

// ---------------------------------------------------------------------------
// Launcher. Inputs: x, freqs_cos, freqs_sin, wq, wk, wv, wo (all fp32).
// Output: (B, Tq, Cc) fp32. Workspace ~80 MB of bf16 intermediates.
// ---------------------------------------------------------------------------
extern "C" void kernel_launch(void* const* d_in, const int* in_sizes, int n_in,
                              void* d_out, int out_size, void* d_ws, size_t ws_size,
                              hipStream_t stream) {
  const float* x  = (const float*)d_in[0];
  const float* fc = (const float*)d_in[1];
  const float* fs = (const float*)d_in[2];
  const float* wq = (const float*)d_in[3];
  const float* wk = (const float*)d_in[4];
  const float* wv = (const float*)d_in[5];
  const float* wo = (const float*)d_in[6];
  float* out = (float*)d_out;

  char* p = (char*)d_ws;
  auto take = [&](size_t elems) { __bf16* r = (__bf16*)p; p += elems * sizeof(__bf16); return r; };
  __bf16* xb  = take((size_t)B_ * Tq * Cc);          // x in bf16
  __bf16* wqT = take((size_t)Cc * Cc);               // wq^T (N x K)
  __bf16* wkT = take((size_t)(NKV * HDm) * Cc);
  __bf16* wvT = take((size_t)(NKV * HDm) * Cc);
  __bf16* woT = take((size_t)Cc * Cc);
  __bf16* Qb  = take((size_t)B_ * Tq * NH  * HDm);   // Q (post-RoPE, in place)
  __bf16* Kb  = take((size_t)B_ * Tq * NKV * HDm);   // K (post-RoPE, in place)
  __bf16* Vb  = take((size_t)B_ * Tq * NKV * HDm);   // V
  __bf16* Vt  = take((size_t)B_ * Tq * NKV * HDm);   // V^T per (b,kv): (HD, T)
  __bf16* Ob  = take((size_t)B_ * Tq * NH  * HDm);   // attention output

  const int M = B_ * Tq;   // 4096

  // 1) conversions
  cvt_f32_bf16<<<(M * Cc / 4 + 255) / 256, 256, 0, stream>>>(x, xb, M * Cc / 4);
  cvtT_f32_bf16<<<dim3(Cc / 32, Cc / 32), 256, 0, stream>>>(wq, wqT, Cc, Cc);
  cvtT_f32_bf16<<<dim3(Cc / 32, (NKV * HDm) / 32), 256, 0, stream>>>(wk, wkT, Cc, NKV * HDm);
  cvtT_f32_bf16<<<dim3(Cc / 32, (NKV * HDm) / 32), 256, 0, stream>>>(wv, wvT, Cc, NKV * HDm);
  cvtT_f32_bf16<<<dim3(Cc / 32, Cc / 32), 256, 0, stream>>>(wo, woT, Cc, Cc);

  // 2) QKV projections (WMMA, TDM-staged tiles)
  gemm_bf16_k<true><<<dim3(M / 128, Cc / 128), 256, 0, stream>>>(xb, wqT, Qb, M, Cc, Cc);
  gemm_bf16_k<true><<<dim3(M / 128, (NKV * HDm) / 128), 256, 0, stream>>>(xb, wkT, Kb, M, NKV * HDm, Cc);
  gemm_bf16_k<true><<<dim3(M / 128, (NKV * HDm) / 128), 256, 0, stream>>>(xb, wvT, Vb, M, NKV * HDm, Cc);

  // 3) RoPE on Q and K (in place)
  {
    int totQ = B_ * Tq * NH * (HDm / 2);
    int totK = B_ * Tq * NKV * (HDm / 2);
    rope_bf16<<<(totQ + 255) / 256, 256, 0, stream>>>(Qb, fc, fs, NH, totQ);
    rope_bf16<<<(totK + 255) / 256, 256, 0, stream>>>(Kb, fc, fs, NKV, totK);
  }

  // 4) V transpose for PV WMMA
  transpose_v<<<dim3(Tq / 32, HDm / 32, B_ * NKV), 256, 0, stream>>>(Vb, Vt);

  // 5) causal GQA flash attention (WMMA)
  attn_flash<<<dim3(Tq / 64, B_ * NH), 128, 0, stream>>>(Qb, Kb, Vt, Ob);

  // 6) output projection, fp32 result (WMMA)
  gemm_bf16_k<false><<<dim3(M / 128, Cc / 128), 256, 0, stream>>>(Ob, woT, out, M, Cc, Cc);
}
